// Plastic_27144193311212
// MI455X (gfx1250) — matrix operand order repaired and checked by reading, not compiled, not run on hardware
//
#include <hip/hip_runtime.h>
#include <hip/hip_fp16.h>

#define IN_F   512
#define OUT_F  512
#define BATCH  512
#define NWG    32          // persistent workgroups, each owns 16 output columns
#define COLS   16          // columns per workgroup (== waves per block)
#define TPB    512         // 16 wave32 waves
#define TRS    513         // padded col-major stride for trace/w/alpha (odd -> bank-conflict-free)
#define ETAS   18          // eta LDS row stride: 16 data dwords + 2 TDM pad dwords

typedef _Float16     v16h  __attribute__((ext_vector_type(16)));
typedef float        v8f   __attribute__((ext_vector_type(8)));
typedef unsigned int u32x4 __attribute__((ext_vector_type(4)));
typedef int          i32x4 __attribute__((ext_vector_type(4)));
typedef int          i32x8 __attribute__((ext_vector_type(8)));

// LDS float offsets (dynamic shared memory; total 43588 floats = 174352 B < 320 KB WGP LDS)
#define OFF_W   0
#define OFF_A   8208        // 16*513
#define OFF_T   16416
#define OFF_E0  24624
#define OFF_E1  33840       // 24624 + 512*18
#define OFF_X   43056
#define OFF_L   43568
#define OFF_R   43584
#define SMEM_FLOATS 43588

// ---- Tensor Data Mover: async 2D tile load of eta[step][:, j0:j0+16] into LDS ----
// D# packed per CDNA5 ISA ch.8: 4B elements, tile 16x512, tensor stride 512,
// pad_enable: 2 dwords every 16 dwords -> LDS row stride 18 (conflict-free).
__device__ __forceinline__ void tdm_issue_eta(const float* eta_g, int step,
                                              unsigned lds_byte_off, int j0) {
  unsigned long long ga =
      (unsigned long long)(const void*)(eta_g + (size_t)step * (IN_F * OUT_F) + j0);
  u32x4 g0;
  g0[0] = 1u;                                              // count=1, user mode
  g0[1] = lds_byte_off;                                    // lds_addr
  g0[2] = (unsigned)(ga & 0xFFFFFFFFull);                  // global_addr lo
  g0[3] = (unsigned)((ga >> 32) & 0x1FFFFFFull) | (2u << 30); // global_addr hi | type=2
  i32x8 g1;
  g1[0] = (int)((2u << 16) | (1u << 20) | (3u << 22) | (1u << 25));
  //              data_size=4B  pad_en    intvl=16dw    pad=2dw
  g1[1] = (int)(512u << 16);       // tensor_dim0 = 512
  g1[2] = 0;                       // dim0 hi, dim1 lo (262144 & 0xffff = 0)
  g1[3] = (int)(4u | (16u << 16)); // dim1 hi (262144>>16), tile_dim0 = 16
  g1[4] = 512;                     // tile_dim1 = 512, tile_dim2 = 0
  g1[5] = 512;                     // tensor_dim0_stride lo
  g1[6] = 0;                       // stride0 hi, stride1 lo
  g1[7] = 4;                       // stride1 hi (262144>>16)
  i32x4 z4 = {0, 0, 0, 0};
  i32x8 z8 = {0, 0, 0, 0, 0, 0, 0, 0};
  __builtin_amdgcn_tensor_load_to_lds(g0, g1, z4, z4, z8, 0);
}

__global__ void __launch_bounds__(TPB)
plastic_scan(const float* __restrict__ X, const float* __restrict__ ETA,
             const float* __restrict__ W, const float* __restrict__ Bp,
             const float* __restrict__ AL, float* __restrict__ OUT,
             unsigned* __restrict__ cnt, float* __restrict__ pmax,
             float* __restrict__ psum) {
  extern __shared__ float sm[];
  float* wl = sm + OFF_W;   // w columns,  [c][i] stride TRS
  float* al = sm + OFF_A;   // alpha cols, [c][i] stride TRS
  float* tr = sm + OFF_T;   // trace cols, [c][i] stride TRS
  float* e0 = sm + OFF_E0;  // eta tile double buffer, [i][c] stride ETAS
  float* e1 = sm + OFF_E1;
  float* xb = sm + OFF_X;   // x_t
  float* lg = sm + OFF_L;   // logits -> x_out
  float* rd = sm + OFF_R;   // {gmax, 1/gsum}

  const int tid  = threadIdx.x;
  const int lane = tid & 31;
  const int wv   = tid >> 5;
  const int g    = blockIdx.x;
  const int j0   = g * COLS;

  const unsigned smbase  = (unsigned)(unsigned long long)sm; // LDS byte offset
  const unsigned etaOff0 = smbase + OFF_E0 * 4u;
  const unsigned etaOff1 = smbase + OFF_E1 * 4u;

  // kick off eta tile for step 0 immediately (overlaps with staging + GEMM)
  if (wv == 0) tdm_issue_eta(ETA, 0, etaOff0, j0);

  // stage w/alpha columns (col-major, padded), zero trace
  for (int idx = tid; idx < COLS * IN_F; idx += TPB) {
    int c = idx & (COLS - 1);
    int i = idx >> 4;
    wl[c * TRS + i] = W[(size_t)i * OUT_F + j0 + c];
    al[c * TRS + i] = AL[(size_t)i * OUT_F + j0 + c];
  }
  for (int idx = tid; idx < COLS * TRS; idx += TPB) tr[idx] = 0.f;
  __syncthreads();

  // ---- phase 0: base logits for ALL steps, OUT[t][j0+c] = (X @ w)[t][j0+c]
  // Real batched GEMM via v_wmma_f32_16x16x32_f16; OUT doubles as scratch
  // (row t is consumed at step t before being overwritten with softmax output).
  const int half_sel = (lane >> 4) & 1;
  const int mrow     = lane & 15;
  for (int mt = wv; mt < BATCH / 16; mt += 16) {
    v8f acc = {};
    const float* xrow = X + (size_t)(mt * 16 + mrow) * IN_F;
    for (int kt = 0; kt < IN_F / 32; ++kt) {
      v16h A, Bm;
      const float* xk = xrow + kt * 32;
#pragma unroll
      for (int e = 0; e < 8; ++e) {            // A: 16x32 f16, ISA lane layout
        A[e]     = (_Float16)xk[half_sel ? 8 + e : e];
        A[8 + e] = (_Float16)xk[16 + (half_sel ? 8 + e : e)];
      }
#pragma unroll
      for (int e = 0; e < 16; ++e) {           // B: 32x16 f16 from w columns in LDS
        Bm[e] = (_Float16)wl[mrow * TRS + kt * 32 + (e & 8) * 2 + half_sel * 8 + (e & 7)];
      }
      acc = __builtin_amdgcn_wmma_f32_16x16x32_f16(false, A, false, Bm,
                                                   (short)0, acc, false, false);
    }
#pragma unroll
    for (int r = 0; r < 8; ++r) {              // D layout: M = r + 8*(lane/16)
      int trow = mt * 16 + r + 8 * half_sel;
      OUT[(size_t)trow * OUT_F + j0 + mrow] = acc[r];
    }
  }
  __threadfence();
  __syncthreads();

  const float bias = Bp[0];  // (softmax-invariant, kept for fidelity)

  // ---- sequential scan over 512 steps ----
  for (int t = 0; t < BATCH; ++t) {
    xb[tid] = X[(size_t)t * IN_F + tid];
    if (wv == 0) {  // prefetch eta tile for t+1 into the other buffer
      int tp = (t + 1 < BATCH) ? (t + 1) : (BATCH - 1);
      tdm_issue_eta(ETA, tp, ((t + 1) & 1) ? etaOff1 : etaOff0, j0);
    }
    __syncthreads();

    // correction matvec: wave wv owns local column wv; unit-stride LDS reads
    {
      const int c = wv;
      float acc = 0.f;
#pragma unroll 4
      for (int k = 0; k < IN_F / 32; ++k) {
        int i = lane + 32 * k;
        acc += xb[i] * al[c * TRS + i] * tr[c * TRS + i];
      }
      acc += __shfl_xor(acc, 16, 32);
      acc += __shfl_xor(acc, 8, 32);
      acc += __shfl_xor(acc, 4, 32);
      acc += __shfl_xor(acc, 2, 32);
      acc += __shfl_xor(acc, 1, 32);
      if (lane == 0)
        lg[c] = OUT[(size_t)t * OUT_F + j0 + c] + bias + acc;
    }
    __syncthreads();

    // cross-WG softmax normalizer: publish (max, sumexp), grid barrier, merge
    if (tid == 0) {
      float lm = -__builtin_inff();
      for (int c = 0; c < COLS; ++c) lm = fmaxf(lm, lg[c]);
      float ls = 0.f;
      for (int c = 0; c < COLS; ++c) ls += __expf(lg[c] - lm);
      int slot = (t & 1) * NWG + g;
      __hip_atomic_store(&pmax[slot], lm, __ATOMIC_RELAXED, __HIP_MEMORY_SCOPE_AGENT);
      __hip_atomic_store(&psum[slot], ls, __ATOMIC_RELAXED, __HIP_MEMORY_SCOPE_AGENT);
      __hip_atomic_fetch_add(cnt, 1u, __ATOMIC_ACQ_REL, __HIP_MEMORY_SCOPE_AGENT);
      unsigned target = (unsigned)NWG * (unsigned)(t + 1);
      while (__hip_atomic_load(cnt, __ATOMIC_ACQUIRE, __HIP_MEMORY_SCOPE_AGENT) < target)
        __builtin_amdgcn_s_sleep(1);
      float gm = -__builtin_inff();
      for (int q = 0; q < NWG; ++q)
        gm = fmaxf(gm, __hip_atomic_load(&pmax[(t & 1) * NWG + q],
                                         __ATOMIC_RELAXED, __HIP_MEMORY_SCOPE_AGENT));
      float gs = 0.f;
      for (int q = 0; q < NWG; ++q) {
        float m = __hip_atomic_load(&pmax[(t & 1) * NWG + q], __ATOMIC_RELAXED,
                                    __HIP_MEMORY_SCOPE_AGENT);
        float s = __hip_atomic_load(&psum[(t & 1) * NWG + q], __ATOMIC_RELAXED,
                                    __HIP_MEMORY_SCOPE_AGENT);
        gs += s * __expf(m - gm);
      }
      rd[0] = gm;
      rd[1] = 1.f / gs;
    }
    if (wv == 0) __builtin_amdgcn_s_wait_tensorcnt(1);  // eta tile t landed
    __syncthreads();

    // x_out for owned columns; overwrite OUT row t (base logits already consumed)
    if (tid < COLS) {
      float xo = __expf(lg[tid] - rd[0]) * rd[1];
      lg[tid]  = xo;
      OUT[(size_t)t * OUT_F + j0 + tid] = xo;
    }
    __syncthreads();

    // trace update: wave wv -> column wv; eta rows at stride 18 (conflict-free)
    {
      const int c  = wv;
      const float xo = lg[c];
      const float* eb = (t & 1) ? e1 : e0;
#pragma unroll 4
      for (int k = 0; k < IN_F / 32; ++k) {
        int i   = lane + 32 * k;
        float e = eb[i * ETAS + c];
        float v = (1.f - e) * tr[c * TRS + i] + e * xb[i] * xo;
        v = fminf(1.f, fmaxf(-1.f, v));
        tr[c * TRS + i] = v;
      }
    }
    __syncthreads();
  }
}

__global__ void init_ws(unsigned* ws) {
  if (threadIdx.x < 256) ws[threadIdx.x] = 0u;
}

extern "C" void kernel_launch(void* const* d_in, const int* in_sizes, int n_in,
                              void* d_out, int out_size, void* d_ws, size_t ws_size,
                              hipStream_t stream) {
  const float* X   = (const float*)d_in[0];
  const float* ETA = (const float*)d_in[1];
  const float* W   = (const float*)d_in[2];
  const float* B   = (const float*)d_in[3];
  const float* AL  = (const float*)d_in[4];
  float* OUT       = (float*)d_out;
  unsigned* ws     = (unsigned*)d_ws;

  init_ws<<<1, 256, 0, stream>>>(ws);  // reset grid-barrier counter each call

  size_t smem = SMEM_FLOATS * sizeof(float);
  plastic_scan<<<NWG, TPB, smem, stream>>>(X, ETA, W, B, AL, OUT, ws,
                                           (float*)(ws + 8), (float*)(ws + 72));
}
